// MultiHeadAttention_56014963474618
// MI455X (gfx1250) — matrix operand order repaired and checked by reading, not compiled, not run on hardware
//
#include <hip/hip_runtime.h>
#include <hip/hip_bf16.h>

// ---------------------------------------------------------------------------
// MI455X / gfx1250 fused multi-head attention layer (bf16 WMMA path)
//   B=8, S=1024, D_MODEL=512, H=8, DK=DV=64
// d_out layout: [0, 4194304) = LayerNorm output, [4194304, +67108864) = attn
// ---------------------------------------------------------------------------

typedef __bf16 bf16;
typedef __attribute__((ext_vector_type(16))) __bf16 bf16x16;
typedef __attribute__((ext_vector_type(8)))  __bf16 bf16x8;
typedef __attribute__((ext_vector_type(8)))  float  f32x8;
typedef __attribute__((ext_vector_type(4)))  float  f32x4;

#define WMMA_BF16(A, B, C) \
  __builtin_amdgcn_wmma_f32_16x16x32_bf16(false, (A), false, (B), (short)0, (C), false, false)

static constexpr int Bb = 8, S = 1024, DM = 512, H = 8, DK = 64;
static constexpr int M_TOK = Bb * S;    // 8192 token rows
static constexpr int SCW = 1028;        // padded LDS score row stride (f32, bank-safe)
static constexpr int BSW = 520;         // padded LDS B-panel row stride (bf16, bank-safe)

// ----------------------- WMMA fragment loaders -----------------------------
// A fragment (16x32, row-major bf16, leading dim ld):
//   lane l: row = l&15; kh = l>>4; elems 0..7 = K kh*8+0..7, 8..15 = K 16+kh*8+0..7
__device__ __forceinline__ bf16x16 load_frag_a(const bf16* __restrict__ p, int ld, int lane) {
  const int r = lane & 15, kh = lane >> 4;
  const bf16* q = p + (size_t)r * ld + kh * 8;
  bf16x8 lo = *reinterpret_cast<const bf16x8*>(q);
  bf16x8 hi = *reinterpret_cast<const bf16x8*>(q + 16);
  bf16x16 a;
#pragma unroll
  for (int i = 0; i < 8; ++i) { a[i] = lo[i]; a[i + 8] = hi[i]; }
  return a;
}

// B fragment (32x16) from an N-major source (row n = column n of B, K contiguous):
//   lane l: col = l&15; kh = l>>4; elems 0..7 = K kh*8+0..7, 8..15 = K 16+kh*8+0..7 (two 16B loads).
__device__ __forceinline__ bf16x16 load_frag_b(const bf16* __restrict__ pT, int ld, int lane) {
  const int n = lane & 15, kq = (lane >> 4) * 8;
  const bf16* q = pT + (size_t)n * ld + kq;
  bf16x8 lo = *reinterpret_cast<const bf16x8*>(q);
  bf16x8 hi = *reinterpret_cast<const bf16x8*>(q + 16);
  bf16x16 b;
#pragma unroll
  for (int i = 0; i < 8; ++i) { b[i] = lo[i]; b[i + 8] = hi[i]; }
  return b;
}

// Same, but via two 16B loads (for LDS panels whose padded stride is 16B-aligned only)
__device__ __forceinline__ bf16x16 load_frag_b16(const bf16* __restrict__ pT, int ld, int lane) {
  const int n = lane & 15, kq = (lane >> 4) * 8;
  const bf16* q = pT + (size_t)n * ld + kq;
  bf16x8 lo = *reinterpret_cast<const bf16x8*>(q);
  bf16x8 hi = *reinterpret_cast<const bf16x8*>(q + 16);
  bf16x16 b;
#pragma unroll
  for (int i = 0; i < 8; ++i) { b[i] = lo[i]; b[i + 8] = hi[i]; }
  return b;
}

__device__ __forceinline__ f32x8 zero_f32x8() {
  f32x8 z;
#pragma unroll
  for (int j = 0; j < 8; ++j) z[j] = 0.0f;
  return z;
}

// ----------------------- prep: f32 -> bf16 inputs ---------------------------
__global__ __launch_bounds__(256) void cvt_inputs_kernel(
    const float* __restrict__ q, const float* __restrict__ k, const float* __restrict__ v,
    bf16* __restrict__ Xq, bf16* __restrict__ Xk, bf16* __restrict__ Xv) {
  const size_t N = (size_t)M_TOK * DM;
  size_t i = (size_t)blockIdx.x * blockDim.x + threadIdx.x;
  if (i < N)            Xq[i]         = (bf16)q[i];
  else if (i < 2 * N)   Xk[i - N]     = (bf16)k[i - N];
  else if (i < 3 * N)   Xv[i - 2 * N] = (bf16)v[i - 2 * N];
}

// ----------------------- prep: transpose+convert weights --------------------
__global__ __launch_bounds__(256) void cvt_weights_kernel(
    const float* __restrict__ WQ, const float* __restrict__ WK,
    const float* __restrict__ WV, const float* __restrict__ Wfc,
    bf16* __restrict__ WqT, bf16* __restrict__ WkT,
    bf16* __restrict__ WvT, bf16* __restrict__ WfcT) {
  int i = blockIdx.x * blockDim.x + threadIdx.x;     // 4 * 512*512
  int which = i >> 18;
  int rem = i & 262143;
  int kk = rem >> 9, n = rem & 511;                  // coalesced source reads
  const float* src = (which == 0) ? WQ : (which == 1) ? WK : (which == 2) ? WV : Wfc;
  bf16* dst = (which == 0) ? WqT : (which == 1) ? WkT : (which == 2) ? WvT : WfcT;
  dst[(size_t)n * DM + kk] = (bf16)src[(size_t)kk * DM + n];   // WT[n][k] = W[k][n]
}

// ----------------------- QKV projection GEMM (WMMA) -------------------------
// grid (64, 8, 3), block 256 (8 waves). B panel (64x512 bf16) staged once in LDS
// and shared by all 8 waves; A fragments double-buffered from global.
__global__ __launch_bounds__(256) void qkv_gemm_kernel(
    const bf16* __restrict__ Xq, const bf16* __restrict__ Xk, const bf16* __restrict__ Xv,
    const bf16* __restrict__ WqT, const bf16* __restrict__ WkT, const bf16* __restrict__ WvT,
    bf16* __restrict__ Qb, bf16* __restrict__ Kb, bf16* __restrict__ Vt) {
  extern __shared__ char dyn_smem[];
  bf16* Bs = (bf16*)dyn_smem;                         // [64][BSW]

  const int lane = threadIdx.x & 31, w = threadIdx.x >> 5;
  const int which = blockIdx.z;
  const bf16* X  = (which == 0) ? Xq  : (which == 1) ? Xk  : Xv;
  const bf16* WT = (which == 0) ? WqT : (which == 1) ? WkT : WvT;
  const int m0 = blockIdx.x * 128 + w * 16;
  const int n0 = blockIdx.y * 64;

  // cooperative B-panel stage: 64 rows x 512 bf16, 16B chunks
  for (int c = threadIdx.x; c < 64 * 64; c += 256) {
    const int n = c >> 6, kc = (c & 63) * 8;
    *reinterpret_cast<bf16x8*>(Bs + n * BSW + kc) =
        *reinterpret_cast<const bf16x8*>(WT + (size_t)(n0 + n) * DM + kc);
  }
  __syncthreads();

  f32x8 acc[4];
#pragma unroll
  for (int t = 0; t < 4; ++t) acc[t] = zero_f32x8();

  const bf16* Abase = X + (size_t)m0 * DM;
  bf16x16 a_cur = load_frag_a(Abase, DM, lane);
#pragma unroll
  for (int k0 = 0; k0 < DM; k0 += 32) {
    bf16x16 a_nxt = (k0 + 32 < DM) ? load_frag_a(Abase + k0 + 32, DM, lane) : a_cur;
#pragma unroll
    for (int t = 0; t < 4; ++t) {
      bf16x16 bf = load_frag_b16(Bs + t * 16 * BSW + k0, BSW, lane);
      acc[t] = WMMA_BF16(a_cur, bf, acc[t]);
    }
    a_cur = a_nxt;
  }

  const int bidx = m0 >> 10;            // batch
  const int s0   = m0 & 1023;           // sequence base (tile stays within one batch)
  if (which == 2) {
    // V stored transposed per head: Vt[b][h][d][s]; consecutive r = consecutive s
#pragma unroll
    for (int t = 0; t < 4; ++t) {
      const int n = n0 + t * 16 + (lane & 15);
      const int h = n >> 6, d = n & 63;
      bf16x8 pk;
#pragma unroll
      for (int r = 0; r < 8; ++r) pk[r] = (bf16)acc[t][r];
      const size_t vbase = (((size_t)bidx * H + h) * DK + d) * S + s0 + ((lane >> 4) * 8);
      *reinterpret_cast<bf16x8*>(Vt + vbase) = pk;
    }
  } else {
    bf16* dst = (which == 0) ? Qb : Kb;
#pragma unroll
    for (int t = 0; t < 4; ++t) {
      const int n = n0 + t * 16 + (lane & 15);
      const int h = n >> 6, d = n & 63;
#pragma unroll
      for (int r = 0; r < 8; ++r) {
        const int s = s0 + ((lane >> 4) * 8) + r;
        dst[(((size_t)bidx * H + h) * S + s) * DK + d] = (bf16)acc[t][r];
      }
    }
  }
}

// ----------------------- fused attention ------------------------------------
// grid (16, H, B), block 128 (4 waves), 257 KB LDS -> 1 block/WGP. Latency is
// hidden with deep in-wave pipelines: K-frags + mask prefetched 3 iterations
// ahead (phase A), V-frags 2 iterations ahead (phase C).
__global__ __launch_bounds__(128) void attn_kernel(
    const bf16* __restrict__ Qb, const bf16* __restrict__ Kb, const bf16* __restrict__ Vt,
    const unsigned char* __restrict__ mask,
    float* __restrict__ attn_out, bf16* __restrict__ Ctx) {
  extern __shared__ char dyn_smem[];
  float* smem = (float*)dyn_smem;
  const int lane = threadIdx.x & 31, w = threadIdx.x >> 5;
  const int qt = blockIdx.x, h = blockIdx.y, b = blockIdx.z;
  const int q0 = qt * 64 + w * 16;

  float* sc   = smem + (size_t)w * 16 * SCW;
  float* winv = smem + (size_t)4 * 16 * SCW + w * 16;

  const bf16* Qbh = Qb + ((size_t)b * H + h) * S * DK;
  const bf16* Kbh = Kb + ((size_t)b * H + h) * S * DK;
  const bf16* Vth = Vt + ((size_t)b * H + h) * DK * S;
  const unsigned char* mb = mask + (size_t)b * S * S;
  float* attn_bh = attn_out + ((size_t)b * H + h) * S * S;

  // ---- Phase A: scores = Q K^T / 8, masked, into LDS ----
  const bf16x16 aq0 = load_frag_a(Qbh + (size_t)q0 * DK + 0,  DK, lane);
  const bf16x16 aq1 = load_frag_a(Qbh + (size_t)q0 * DK + 32, DK, lane);

  bf16x16 kb0[4], kb1[4];
  unsigned char mk[4][8];
#pragma unroll
  for (int p = 0; p < 3; ++p) {                       // preload pipeline depth 3
    kb0[p] = load_frag_b(Kbh + (size_t)(p * 16) * DK + 0,  DK, lane);
    kb1[p] = load_frag_b(Kbh + (size_t)(p * 16) * DK + 32, DK, lane);
    const int pcol = p * 16 + (lane & 15);
#pragma unroll
    for (int r = 0; r < 8; ++r)
      mk[p][r] = mb[(size_t)(q0 + (lane >> 4) * 8 + r) * S + pcol];
  }

#pragma unroll
  for (int nt = 0; nt < 64; ++nt) {
    const int cs = nt & 3;
    const int pn = nt + 3, ps = pn & 3;
    if (pn < 64) {
      kb0[ps] = load_frag_b(Kbh + (size_t)(pn * 16) * DK + 0,  DK, lane);
      kb1[ps] = load_frag_b(Kbh + (size_t)(pn * 16) * DK + 32, DK, lane);
      const int pcol = pn * 16 + (lane & 15);
#pragma unroll
      for (int r = 0; r < 8; ++r)
        mk[ps][r] = mb[(size_t)(q0 + (lane >> 4) * 8 + r) * S + pcol];
    }
    f32x8 acc = zero_f32x8();
    acc = WMMA_BF16(aq0, kb0[cs], acc);
    acc = WMMA_BF16(aq1, kb1[cs], acc);
    const int col = nt * 16 + (lane & 15);
#pragma unroll
    for (int r = 0; r < 8; ++r) {
      const int row = (lane >> 4) * 8 + r;
      const float v = mk[cs][r] ? -1e9f : acc[r] * 0.125f;   // 1/sqrt(64)
      sc[row * SCW + col] = v;
    }
  }
  asm volatile("s_wait_dscnt 0" ::: "memory");

  // ---- Phase B: safe softmax, float4-vectorized (2 lanes per row) ----
  {
    const int row = lane & 15, half = lane >> 4;
    f32x4* rp4 = reinterpret_cast<f32x4*>(sc + row * SCW + half * 512);
    float m = -3.0e38f;
    for (int j = 0; j < 128; ++j) {
      f32x4 v = rp4[j];
      m = fmaxf(m, fmaxf(fmaxf(v[0], v[1]), fmaxf(v[2], v[3])));
    }
    m = fmaxf(m, __shfl_xor(m, 16));
    float ssum = 0.0f;
    for (int j = 0; j < 128; ++j) {
      f32x4 v = rp4[j];
      v[0] = __expf(v[0] - m); v[1] = __expf(v[1] - m);
      v[2] = __expf(v[2] - m); v[3] = __expf(v[3] - m);
      rp4[j] = v;
      ssum += (v[0] + v[1]) + (v[2] + v[3]);
    }
    ssum += __shfl_xor(ssum, 16);
    if (half == 0) winv[row] = 1.0f / ssum;
  }
  asm volatile("s_wait_dscnt 0" ::: "memory");

  // ---- Phase C: normalize, emit attn (nontemporal, single HBM write), P*V ----
  f32x8 acc[4];
#pragma unroll
  for (int t = 0; t < 4; ++t) acc[t] = zero_f32x8();
  const float inv = winv[lane & 15];
  const int row = lane & 15, kh = lane >> 4;

  bf16x16 vb[3][4];                                   // distance-2 V prefetch ring
#pragma unroll
  for (int p = 0; p < 2; ++p)
#pragma unroll
    for (int dt = 0; dt < 4; ++dt)
      vb[p][dt] = load_frag_b(Vth + (size_t)(dt * 16) * S + p * 32, S, lane);

#pragma unroll
  for (int st = 0; st < 32; ++st) {
    const int cs = st % 3;
    const int pn = st + 2, ps = pn % 3;
    if (pn < 32) {
#pragma unroll
      for (int dt = 0; dt < 4; ++dt)
        vb[ps][dt] = load_frag_b(Vth + (size_t)(dt * 16) * S + pn * 32, S, lane);
    }

    const int k0 = st * 32;
    const f32x4* rp4 = reinterpret_cast<const f32x4*>(sc + row * SCW + k0 + kh * 8);
    f32x4 p0 = rp4[0] * inv, p1 = rp4[1] * inv;
    f32x4 p2 = rp4[4] * inv, p3 = rp4[5] * inv;

    f32x4* op4 = reinterpret_cast<f32x4*>(attn_bh + (size_t)(q0 + row) * S + k0 + kh * 8);
    __builtin_nontemporal_store(p0, op4 + 0);
    __builtin_nontemporal_store(p1, op4 + 1);
    __builtin_nontemporal_store(p2, op4 + 4);
    __builtin_nontemporal_store(p3, op4 + 5);

    bf16x16 a;
#pragma unroll
    for (int i = 0; i < 4; ++i) {
      a[i]      = (bf16)p0[i];
      a[i + 4]  = (bf16)p1[i];
      a[i + 8]  = (bf16)p2[i];
      a[i + 12] = (bf16)p3[i];
    }
#pragma unroll
    for (int dt = 0; dt < 4; ++dt) acc[dt] = WMMA_BF16(a, vb[cs][dt], acc[dt]);
  }

  // context -> Ctx[b][s][h*64+d] (bf16, token-major for the FC GEMM A-fragments)
#pragma unroll
  for (int dt = 0; dt < 4; ++dt) {
    const int c = h * DK + dt * 16 + (lane & 15);
#pragma unroll
    for (int r = 0; r < 8; ++r) {
      const int s = q0 + (lane >> 4) * 8 + r;
      Ctx[((size_t)b * S + s) * DM + c] = (bf16)acc[dt][r];
    }
  }
}

// ----------------------- output projection GEMM -----------------------------
__global__ __launch_bounds__(256) void fc_gemm_kernel(
    const bf16* __restrict__ Ctx, const bf16* __restrict__ WfcT,
    float* __restrict__ OutPre) {
  extern __shared__ char dyn_smem[];
  bf16* Bs = (bf16*)dyn_smem;                         // [64][BSW]
  const int lane = threadIdx.x & 31, w = threadIdx.x >> 5;
  const int m0 = blockIdx.x * 128 + w * 16;
  const int n0 = blockIdx.y * 64;

  for (int c = threadIdx.x; c < 64 * 64; c += 256) {
    const int n = c >> 6, kc = (c & 63) * 8;
    *reinterpret_cast<bf16x8*>(Bs + n * BSW + kc) =
        *reinterpret_cast<const bf16x8*>(WfcT + (size_t)(n0 + n) * DM + kc);
  }
  __syncthreads();

  f32x8 acc[4];
#pragma unroll
  for (int t = 0; t < 4; ++t) acc[t] = zero_f32x8();

  const bf16* Abase = Ctx + (size_t)m0 * DM;
  bf16x16 a_cur = load_frag_a(Abase, DM, lane);
#pragma unroll
  for (int k0 = 0; k0 < DM; k0 += 32) {
    bf16x16 a_nxt = (k0 + 32 < DM) ? load_frag_a(Abase + k0 + 32, DM, lane) : a_cur;
#pragma unroll
    for (int t = 0; t < 4; ++t) {
      bf16x16 bf = load_frag_b16(Bs + t * 16 * BSW + k0, BSW, lane);
      acc[t] = WMMA_BF16(a_cur, bf, acc[t]);
    }
    a_cur = a_nxt;
  }
#pragma unroll
  for (int t = 0; t < 4; ++t) {
    const int n = n0 + t * 16 + (lane & 15);
#pragma unroll
    for (int r = 0; r < 8; ++r) {
      const int m = m0 + (lane >> 4) * 8 + r;
      OutPre[(size_t)m * DM + n] = acc[t][r];
    }
  }
}

// ----------------------- residual + LayerNorm -------------------------------
__global__ __launch_bounds__(256) void ln_kernel(
    const float* __restrict__ OutPre, const float* __restrict__ inQ,
    const float* __restrict__ gamma, const float* __restrict__ beta,
    float* __restrict__ out) {
  const int lane = threadIdx.x & 31, w = threadIdx.x >> 5;
  const int row = blockIdx.x * 8 + w;
  const int c0 = lane * 16;
  const f32x4* xp = reinterpret_cast<const f32x4*>(OutPre + (size_t)row * DM + c0);
  const f32x4* rq = reinterpret_cast<const f32x4*>(inQ    + (size_t)row * DM + c0);

  float x[16], sum = 0.0f, sq = 0.0f;
#pragma unroll
  for (int i = 0; i < 4; ++i) {
    f32x4 a = xp[i], b = rq[i];
#pragma unroll
    for (int j = 0; j < 4; ++j) x[4 * i + j] = a[j] + b[j];
  }
#pragma unroll
  for (int i = 0; i < 16; ++i) { sum += x[i]; sq += x[i] * x[i]; }
#pragma unroll
  for (int m = 16; m >= 1; m >>= 1) {
    sum += __shfl_xor(sum, m);
    sq  += __shfl_xor(sq,  m);
  }
  const float mu  = sum * (1.0f / 512.0f);
  const float var = sq * (1.0f / 512.0f) - mu * mu;
  const float rs  = rsqrtf(var + 1e-5f);

  const f32x4* g4 = reinterpret_cast<const f32x4*>(gamma + c0);
  const f32x4* b4 = reinterpret_cast<const f32x4*>(beta + c0);
  f32x4* o4 = reinterpret_cast<f32x4*>(out + (size_t)row * DM + c0);
#pragma unroll
  for (int i = 0; i < 4; ++i) {
    f32x4 g = g4[i], bt = b4[i], o;
#pragma unroll
    for (int j = 0; j < 4; ++j) o[j] = (x[4 * i + j] - mu) * rs * g[j] + bt[j];
    o4[i] = o;
  }
}

// ----------------------- launcher -------------------------------------------
extern "C" void kernel_launch(void* const* d_in, const int* in_sizes, int n_in,
                              void* d_out, int out_size, void* d_ws, size_t ws_size,
                              hipStream_t stream) {
  const float* inQ  = (const float*)d_in[0];
  const float* inK  = (const float*)d_in[1];
  const float* inV  = (const float*)d_in[2];
  const unsigned char* mask = (const unsigned char*)d_in[3];   // jnp bool = 1 byte
  const float* WQ   = (const float*)d_in[4];
  const float* WK   = (const float*)d_in[5];
  const float* WV   = (const float*)d_in[6];
  const float* Wfc  = (const float*)d_in[7];
  const float* gam  = (const float*)d_in[8];
  const float* bet  = (const float*)d_in[9];

  // workspace carve-up (~74 MB, all regions well aligned)
  constexpr size_t NTOK = (size_t)M_TOK * DM;           // 4,194,304 elements
  constexpr size_t NW   = (size_t)DM * DM;              //   262,144 elements
  char* ws = (char*)d_ws;
  size_t off = 0;
  bf16* Xq   = (bf16*)(ws + off); off += NTOK * 2;
  bf16* Xk   = (bf16*)(ws + off); off += NTOK * 2;
  bf16* Xv   = (bf16*)(ws + off); off += NTOK * 2;
  bf16* WqT  = (bf16*)(ws + off); off += NW * 2;
  bf16* WkT  = (bf16*)(ws + off); off += NW * 2;
  bf16* WvT  = (bf16*)(ws + off); off += NW * 2;
  bf16* WfcT = (bf16*)(ws + off); off += NW * 2;
  bf16* Qb   = (bf16*)(ws + off); off += NTOK * 2;
  bf16* Kb   = (bf16*)(ws + off); off += NTOK * 2;
  bf16* Vt   = (bf16*)(ws + off); off += NTOK * 2;
  bf16* Ctx  = (bf16*)(ws + off); off += NTOK * 2;
  float* OutPre = (float*)(ws + off); off += NTOK * 4;
  (void)off; (void)ws_size; (void)in_sizes; (void)n_in; (void)out_size;

  float* outLN   = (float*)d_out;                        // [8192][512]
  float* outAttn = (float*)d_out + NTOK;                 // [B][H][S][S]

  const size_t gemm_smem = (size_t)64 * BSW * sizeof(bf16);              // 66,560 B
  const size_t att_smem  = ((size_t)4 * 16 * SCW + 64) * sizeof(float);  // 263,424 B < 320 KB

  // 1) f32 -> bf16 inputs
  cvt_inputs_kernel<<<(unsigned)((3 * NTOK) / 256), 256, 0, stream>>>(inQ, inK, inV, Xq, Xk, Xv);
  // 2) transpose + convert weights
  cvt_weights_kernel<<<(unsigned)((4 * NW) / 256), 256, 0, stream>>>(WQ, WK, WV, Wfc, WqT, WkT, WvT, WfcT);
  // 3) Q/K/V projections (V stored transposed per head)
  qkv_gemm_kernel<<<dim3(M_TOK / 128, DM / 64, 3), 256, gemm_smem, stream>>>(
      Xq, Xk, Xv, WqT, WkT, WvT, Qb, Kb, Vt);
  // 4) fused attention: scores -> mask -> softmax -> attn write -> P*V
  attn_kernel<<<dim3(S / 64, H, Bb), 128, att_smem, stream>>>(Qb, Kb, Vt, mask, outAttn, Ctx);
  // 5) output projection
  fc_gemm_kernel<<<dim3(M_TOK / 128, DM / 64), 256, gemm_smem, stream>>>(Ctx, WfcT, OutPre);
  // 6) residual + LayerNorm
  ln_kernel<<<M_TOK / 8, 256, 0, stream>>>(OutPre, inQ, gam, bet, outLN);
}
